// LinearTransformerFlow_88115549044865
// MI455X (gfx1250) — compile-verified, matrix operand order checked by reading, not run on hardware
//
#include <hip/hip_runtime.h>

typedef __attribute__((ext_vector_type(16))) _Float16 v16h;
typedef __attribute__((ext_vector_type(8)))  float    v8f;
typedef __attribute__((ext_vector_type(4)))  int      v4i;

#define N2T   1024
#define BATCH 128
#define DM    256
#define HEADS 4
#define DH    64
#define MROWS (BATCH * N2T)   /* 131072 token rows */

/* ---- WMMA GEMM tiling ---- */
#define BM 128
#define BK 32
#define LDP (BK + 8)          /* padded LDS pitch (halves): 80B stride, conflict-free b128 reads */

enum { ACT_NONE = 0, ACT_ELU1 = 1, ACT_RELU = 2, ACT_LEAKY = 3 };

/* CDNA5 async global->LDS path (ASYNCcnt), with sync fallback */
#if defined(__has_builtin)
# if __has_builtin(__builtin_amdgcn_global_load_async_to_lds_b128) && \
     __has_builtin(__builtin_amdgcn_s_wait_asynccnt)
#  define USE_ASYNC_LDS 1
# endif
#endif
#ifndef USE_ASYNC_LDS
# define USE_ASYNC_LDS 0
#endif

#define AS1 __attribute__((address_space(1)))
#define AS3 __attribute__((address_space(3)))

/* ------------------------------------------------------------------ */
/* src[b,t,d] = emb2[t,d] * x_flat[t*128+b]; writes f32 + f16 copies  */
/* ------------------------------------------------------------------ */
__global__ __launch_bounds__(256)
void build_src_kernel(const float* __restrict__ x, const float* __restrict__ embed,
                      float* __restrict__ srcF, _Float16* __restrict__ srcH) {
  size_t idx = (size_t)blockIdx.x * 256 + threadIdx.x;
  int d = (int)(idx % DM);
  size_t bt = idx / DM;
  int t = (int)(bt % N2T);
  int b = (int)(bt / N2T);
  float e = (d < 128) ? embed[(size_t)(t >> 5) * 128 + d]
                      : embed[(size_t)((t & 31) + 32) * 128 + (d - 128)];
  float v = e * x[(size_t)t * BATCH + b];
  srcF[idx] = v;
  srcH[idx] = (_Float16)v;
}

/* ------------------------------------------------------------------ */
/* Y = act(A @ W^T + bias) (+ res); A: MxK f16, W: NxK f16 row-major. */
/* Wave32 WMMA f32_16x16x32_f16; 8 waves (4x2); wave tile 32 x BNT/2. */
/* Double-buffered LDS; async global->LDS staging when available.     */
/* Fragment layouts per CDNA5 ISA §7.12.2.                 */
/* ------------------------------------------------------------------ */
template<int BNT, int ACT, bool RES, bool WF, bool WH>
__global__ __launch_bounds__(256)
void gemm_kernel(const _Float16* __restrict__ A, const _Float16* __restrict__ W,
                 const float* __restrict__ bias, const float* __restrict__ res,
                 float* __restrict__ outF, _Float16* __restrict__ outH, int N) {
  constexpr int CN    = BNT / 32;   /* 16-wide N tiles per wave */
  constexpr int B_OPS = BNT / 64;   /* 16B staging ops per thread for W tile */
  constexpr int NOPS  = 2 + B_OPS;  /* async ops issued per thread per tile */
  constexpr int K     = DM;         /* all GEMMs here have K = 256 */

  __shared__ _Float16 sA[2][BM][LDP];
  __shared__ _Float16 sB[2][BNT][LDP];

  const int tid  = threadIdx.x;
  const int lane = tid & 31;
  const int wave = tid >> 5;
  const int wm   = wave >> 1;       /* 0..3 */
  const int wn   = wave & 1;        /* 0..1 */
  const int l16  = lane & 15;
  const int kg   = lane >> 4;       /* half-wave K-phase selector */
  const size_t bm = (size_t)blockIdx.x * BM;
  const int    bn = blockIdx.y * BNT;

  auto stage = [&](int buf, int kk) {
#pragma unroll
    for (int it = 0; it < 2; ++it) {          /* A tile: 512 x 16B chunks */
      int idx = tid + it * 256;
      int row = idx >> 2, ch = idx & 3;
      const _Float16* g = A + (bm + row) * (size_t)K + kk + ch * 8;
      _Float16* l = &sA[buf][row][ch * 8];
#if USE_ASYNC_LDS
      __builtin_amdgcn_global_load_async_to_lds_b128(
          (AS1 v4i*)(size_t)g, (AS3 v4i*)(size_t)l, 0, 0);
#else
      *(uint4*)l = *(const uint4*)g;
#endif
    }
#pragma unroll
    for (int it = 0; it < B_OPS; ++it) {      /* W tile: BNT*4 x 16B chunks */
      int idx = tid + it * 256;
      int row = idx >> 2, ch = idx & 3;
      const _Float16* g = W + (size_t)(bn + row) * K + kk + ch * 8;
      _Float16* l = &sB[buf][row][ch * 8];
#if USE_ASYNC_LDS
      __builtin_amdgcn_global_load_async_to_lds_b128(
          (AS1 v4i*)(size_t)g, (AS3 v4i*)(size_t)l, 0, 0);
#else
      *(uint4*)l = *(const uint4*)g;
#endif
    }
  };

  v8f acc[2][CN] = {};

  stage(0, 0);
  constexpr int NK = K / BK;
  for (int i = 0; i < NK; ++i) {
    int buf = i & 1;
    if (i + 1 < NK) {
      stage(buf ^ 1, (i + 1) * BK);
#if USE_ASYNC_LDS
      __builtin_amdgcn_s_wait_asynccnt(NOPS);  /* retire current tile; next stays in flight */
#endif
    } else {
#if USE_ASYNC_LDS
      __builtin_amdgcn_s_wait_asynccnt(0);
#endif
    }
    __syncthreads();

    v16h af[2], bf[CN];
#pragma unroll
    for (int r = 0; r < 2; ++r) {
      int row = wm * 32 + r * 16 + l16;
      uint4 lo = *(const uint4*)&sA[buf][row][kg * 8];
      uint4 hi = *(const uint4*)&sA[buf][row][kg * 8 + 16];
      ((uint4*)&af[r])[0] = lo;
      ((uint4*)&af[r])[1] = hi;
    }
#pragma unroll
    for (int c = 0; c < CN; ++c) {
      int col = wn * (BNT / 2) + c * 16 + l16;
      uint4 lo = *(const uint4*)&sB[buf][col][kg * 8];
      uint4 hi = *(const uint4*)&sB[buf][col][kg * 8 + 16];
      ((uint4*)&bf[c])[0] = lo;
      ((uint4*)&bf[c])[1] = hi;
    }
#pragma unroll
    for (int r = 0; r < 2; ++r)
#pragma unroll
      for (int c = 0; c < CN; ++c)
        acc[r][c] = __builtin_amdgcn_wmma_f32_16x16x32_f16(
            false, af[r], false, bf[c], (short)0, acc[r][c], false, false);
    __syncthreads();
  }

  /* epilogue: bias + activation + residual, f32/f16 stores */
#pragma unroll
  for (int r = 0; r < 2; ++r) {
#pragma unroll
    for (int c = 0; c < CN; ++c) {
      int colg = bn + wn * (BNT / 2) + c * 16 + l16;
      float bv = bias[colg];
#pragma unroll
      for (int e = 0; e < 8; ++e) {
        size_t rowg = bm + wm * 32 + r * 16 + e + kg * 8;
        float v = acc[r][c][e] + bv;
        if (ACT == ACT_ELU1)       v = (v > 0.f) ? (v + 1.f) : __expf(v);
        else if (ACT == ACT_RELU)  v = fmaxf(v, 0.f);
        else if (ACT == ACT_LEAKY) v = (v > 0.f) ? v : 0.01f * v;
        size_t oidx = rowg * (size_t)N + colg;
        if (RES) v += res[oidx];
        if (WF)  outF[oidx] = v;
        if (WH)  outH[oidx] = (_Float16)v;
      }
    }
  }
}

/* Ksum[b,d] = sum_l Kf[b,l,d] */
__global__ __launch_bounds__(256)
void ksum_kernel(const float* __restrict__ Kf, float* __restrict__ ks) {
  int b = blockIdx.x, d = threadIdx.x;
  const float* p = Kf + (size_t)b * N2T * DM + d;
  float s = 0.f;
  for (int l = 0; l < N2T; ++l) s += p[(size_t)l * DM];
  ks[b * DM + d] = s;
}

/* KV[b,h,d,m] = sum_s Kf[b,s,h*64+d] * V[b,s,h*64+m] */
__global__ __launch_bounds__(256)
void kv_kernel(const float* __restrict__ Kf, const float* __restrict__ V,
               float* __restrict__ KV) {
  int bh = blockIdx.x;
  int b = bh >> 2, h = bh & 3;
  __shared__ float sk[8][64], sv[8][64];
  float acc[16];
#pragma unroll
  for (int i = 0; i < 16; ++i) acc[i] = 0.f;
  int m = threadIdx.x & 63, dg = threadIdx.x >> 6;
  for (int s0 = 0; s0 < N2T; s0 += 8) {
#pragma unroll
    for (int j = 0; j < 4; ++j) {
      int ii = threadIdx.x + j * 256;
      int r = ii >> 7, hf = (ii >> 6) & 1, e = ii & 63;
      size_t off = ((size_t)b * N2T + s0 + r) * DM + h * DH + e;
      if (hf) sv[r][e] = V[off]; else sk[r][e] = Kf[off];
    }
    __syncthreads();
#pragma unroll
    for (int r = 0; r < 8; ++r) {
      float vv = sv[r][m];
#pragma unroll
      for (int i = 0; i < 16; ++i) acc[i] += sk[r][dg * 16 + i] * vv;
    }
    __syncthreads();
  }
#pragma unroll
  for (int i = 0; i < 16; ++i)
    KV[((size_t)bh * DH + dg * 16 + i) * DH + m] = acc[i];
}

/* attn[b,l,h*64+m] = (Qf . KV) / (Qf . Ksum + 1e-6), f16 out for O-GEMM */
__global__ __launch_bounds__(64)
void attn_kernel(const float* __restrict__ Qf, const float* __restrict__ KV,
                 const float* __restrict__ ks, _Float16* __restrict__ attnH) {
  int id = blockIdx.x;           /* b*64 + lc*4 + h */
  int h = id & 3, lc = (id >> 2) & 15, b = id >> 6;
  __shared__ float skv[64][64];
  __shared__ float sks[64];
  __shared__ float sq[64];
  int m = threadIdx.x;
#pragma unroll 4
  for (int d = 0; d < 64; ++d)
    skv[d][m] = KV[(((size_t)(b * 4 + h)) * 64 + d) * 64 + m];
  sks[m] = ks[b * DM + h * DH + m];
  __syncthreads();
  for (int l = lc * 64; l < lc * 64 + 64; ++l) {
    size_t row = (size_t)b * N2T + l;
    sq[m] = Qf[row * DM + h * DH + m];
    __syncthreads();
    float z = 1e-6f, a = 0.f;
#pragma unroll 8
    for (int d = 0; d < 64; ++d) { float q = sq[d]; z += q * sks[d]; a += q * skv[d][m]; }
    attnH[row * DM + h * DH + m] = (_Float16)(a / z);
    __syncthreads();
  }
}

/* row LayerNorm over 256; optional f32 and f16 outputs */
__global__ __launch_bounds__(256)
void ln_kernel(const float* __restrict__ in, const float* __restrict__ g,
               const float* __restrict__ bb, float* __restrict__ outF,
               _Float16* __restrict__ outH) {
  __shared__ float red[256];
  size_t row = blockIdx.x;
  int tid = threadIdx.x;
  float x = in[row * DM + tid];
  red[tid] = x; __syncthreads();
  for (int o = 128; o > 0; o >>= 1) { if (tid < o) red[tid] += red[tid + o]; __syncthreads(); }
  float mean = red[0] * (1.f / DM);
  __syncthreads();
  float d = x - mean;
  red[tid] = d * d; __syncthreads();
  for (int o = 128; o > 0; o >>= 1) { if (tid < o) red[tid] += red[tid + o]; __syncthreads(); }
  float var = red[0] * (1.f / DM);
  float y = d * rsqrtf(var + 1e-5f) * g[tid] + bb[tid];
  if (outF) outF[row * DM + tid] = y;
  if (outH) outH[row * DM + tid] = (_Float16)y;
}

/* mean over L */
__global__ __launch_bounds__(256)
void meanl_kernel(const float* __restrict__ in, float* __restrict__ out) {
  int b = blockIdx.x, d = threadIdx.x;
  float s = 0.f;
  for (int l = 0; l < N2T; ++l) s += in[((size_t)b * N2T + l) * DM + d];
  out[b * DM + d] = s * (1.f / N2T);
}

/* stop head: leaky(mean@Wh1^T+b) @ Wh2^T + b */
__global__ __launch_bounds__(64)
void stop_mlp_kernel(const float* __restrict__ mean, const float* __restrict__ Wh1,
                     const float* __restrict__ bh1, const float* __restrict__ Wh2,
                     const float* __restrict__ bh2, float* __restrict__ stp) {
  __shared__ float sm[256];
  __shared__ float sh[64];
  int b = blockIdx.x, tid = threadIdx.x;
#pragma unroll
  for (int j = 0; j < 4; ++j) sm[tid + j * 64] = mean[b * DM + tid + j * 64];
  __syncthreads();
  float a = bh1[tid];
  for (int i = 0; i < 256; ++i) a += Wh1[tid * 256 + i] * sm[i];
  a = (a > 0.f) ? a : 0.01f * a;
  sh[tid] = a; __syncthreads();
  if (tid == 0) {
    float s = bh2[0];
    for (int i = 0; i < 64; ++i) s += sh[i] * Wh2[i];
    stp[b] = s;
  }
}

/* logits[r] = dot(h[r, :64], w2) + b2 ; 8 lanes/row, shuffle reduce */
__global__ __launch_bounds__(256)
void dot64_kernel(const float* __restrict__ hbuf, const float* __restrict__ w2,
                  const float* __restrict__ b2, float* __restrict__ out) {
  int r = blockIdx.x * 32 + (threadIdx.x >> 3);
  int j = threadIdx.x & 7;
  float s = 0.f;
#pragma unroll
  for (int i = 0; i < 8; ++i) s += hbuf[(size_t)r * 64 + j * 8 + i] * w2[j * 8 + i];
  s += __shfl_xor(s, 4);
  s += __shfl_xor(s, 2);
  s += __shfl_xor(s, 1);
  if (j == 0) out[r] = s + b2[0];
}

/* masked log-softmax over n (<=1025) per batch row */
__global__ __launch_bounds__(256)
void lsm_kernel(const float* __restrict__ act, const float* __restrict__ stp,
                const int* __restrict__ mask, float* __restrict__ out, int n) {
  __shared__ float sx[1025];
  __shared__ float red[256];
  int b = blockIdx.x, tid = threadIdx.x;
  for (int i = tid; i < n; i += 256) {
    float v = (stp && i == n - 1) ? stp[b] : act[(size_t)b * N2T + i];
    float mk = (float)mask[(size_t)b * n + i];
    sx[i] = v - (1.f - mk) * 1e8f;
  }
  __syncthreads();
  float mx = -3.4e38f;
  for (int i = tid; i < n; i += 256) mx = fmaxf(mx, sx[i]);
  red[tid] = mx; __syncthreads();
  for (int o = 128; o > 0; o >>= 1) { if (tid < o) red[tid] = fmaxf(red[tid], red[tid + o]); __syncthreads(); }
  mx = red[0]; __syncthreads();
  float s = 0.f;
  for (int i = tid; i < n; i += 256) s += __expf(sx[i] - mx);
  red[tid] = s; __syncthreads();
  for (int o = 128; o > 0; o >>= 1) { if (tid < o) red[tid] += red[tid + o]; __syncthreads(); }
  float lse = mx + logf(red[0]);
  for (int i = tid; i < n; i += 256) out[(size_t)b * n + i] = sx[i] - lse;
}

__global__ __launch_bounds__(256)
void f2h_kernel(const float* __restrict__ in, _Float16* __restrict__ out, int n) {
  int i = blockIdx.x * 256 + threadIdx.x;
  if (i < n) out[i] = (_Float16)in[i];
}

/* ------------------------------------------------------------------ */
struct EncW {
  const float *Wq,*bq,*Wk,*bk,*Wv,*bv,*Wo,*bo,*W1,*f1,*W2,*f2,*g1,*c1,*g2,*c2,*gf,*cf;
};

static void fillEncIns(void* const* d, int b, EncW& e) {
  e.Wq=(const float*)d[b+0];  e.bq=(const float*)d[b+1];
  e.Wk=(const float*)d[b+2];  e.bk=(const float*)d[b+3];
  e.Wv=(const float*)d[b+4];  e.bv=(const float*)d[b+5];
  e.Wo=(const float*)d[b+6];  e.bo=(const float*)d[b+7];
  e.W1=(const float*)d[b+8];  e.f1=(const float*)d[b+9];
  e.W2=(const float*)d[b+10]; e.f2=(const float*)d[b+11];
  e.g1=(const float*)d[b+12]; e.c1=(const float*)d[b+13];
  e.g2=(const float*)d[b+14]; e.c2=(const float*)d[b+15];
  e.gf=(const float*)d[b+16]; e.cf=(const float*)d[b+17];
}
static void fillEncAlpha(void* const* d, int b, EncW& e) {
  e.W1=(const float*)d[b+0];  e.W2=(const float*)d[b+1];
  e.Wk=(const float*)d[b+2];  e.Wo=(const float*)d[b+3];
  e.Wq=(const float*)d[b+4];  e.Wv=(const float*)d[b+5];
  e.bk=(const float*)d[b+6];  e.bo=(const float*)d[b+7];
  e.bq=(const float*)d[b+8];  e.bv=(const float*)d[b+9];
  e.c1=(const float*)d[b+10]; e.c2=(const float*)d[b+11];
  e.cf=(const float*)d[b+12]; e.f1=(const float*)d[b+13];
  e.f2=(const float*)d[b+14]; e.g1=(const float*)d[b+15];
  e.g2=(const float*)d[b+16]; e.gf=(const float*)d[b+17];
}

extern "C" void kernel_launch(void* const* d_in, const int* in_sizes, int n_in,
                              void* d_out, int out_size, void* d_ws, size_t ws_size,
                              hipStream_t stream) {
  /* -------- input mapping (insertion-order vs tree-flatten alphabetical) -------- */
  const float *x, *embed;
  const int *fmask, *bmask;
  EncW rep, fwd, stop, bwd;
  const float *fwdWh1,*fwdbh1,*fwdWh2,*fwdbh2;
  const float *stopWh1,*stopbh1,*stopWh2,*stopbh2;
  const float *bwdWh1,*bwdbh1,*bwdWh2,*bwdbh2;

  if (in_sizes[3] == 2 * N2T * 128) {        /* insertion order: embed at index 3 */
    x     = (const float*)d_in[0];
    fmask = (const int*)d_in[1];
    bmask = (const int*)d_in[2];
    embed = (const float*)d_in[3];
    fillEncIns(d_in, 4,  rep);
    fillEncIns(d_in, 22, fwd);
    fillEncIns(d_in, 40, stop);
    fillEncIns(d_in, 58, bwd);
    fwdWh1=(const float*)d_in[76]; fwdbh1=(const float*)d_in[77];
    fwdWh2=(const float*)d_in[78]; fwdbh2=(const float*)d_in[79];
    stopWh1=(const float*)d_in[80]; stopbh1=(const float*)d_in[81];
    stopWh2=(const float*)d_in[82]; stopbh2=(const float*)d_in[83];
    bwdWh1=(const float*)d_in[84]; bwdbh1=(const float*)d_in[85];
    bwdWh2=(const float*)d_in[86]; bwdbh2=(const float*)d_in[87];
  } else {                                    /* alphabetical tree-flatten */
    bmask = (const int*)d_in[0];
    fmask = (const int*)d_in[1];
    fillEncAlpha(d_in, 2, bwd);
    bwdWh1=(const float*)d_in[20]; bwdWh2=(const float*)d_in[21];
    bwdbh1=(const float*)d_in[22]; bwdbh2=(const float*)d_in[23];
    embed = (const float*)d_in[24];
    fillEncAlpha(d_in, 25, fwd);
    fwdWh1=(const float*)d_in[43]; fwdWh2=(const float*)d_in[44];
    fwdbh1=(const float*)d_in[45]; fwdbh2=(const float*)d_in[46];
    fillEncAlpha(d_in, 47, rep);
    fillEncAlpha(d_in, 65, stop);
    stopWh1=(const float*)d_in[83]; stopWh2=(const float*)d_in[84];
    stopbh1=(const float*)d_in[85]; stopbh2=(const float*)d_in[86];
    x = (const float*)d_in[87];
  }

  /* -------- workspace carve -------- */
  char* wp = (char*)d_ws;
  auto carve = [&](size_t bytes) -> void* {
    void* p = (void*)wp;
    wp += (bytes + 255) & ~(size_t)255;
    return p;
  };
  const size_t S = (size_t)MROWS * DM;
  float*    SRCF = (float*)carve(S * 4);
  _Float16* SRCH = (_Float16*)carve(S * 2);
  float*    REPF = (float*)carve(S * 4);
  _Float16* REPH = (_Float16*)carve(S * 2);
  float*    B0   = (float*)carve(S * 4);
  float*    B1   = (float*)carve(S * 4);
  float*    B2   = (float*)carve(S * 4);
  _Float16* H0   = (_Float16*)carve(S * 2);
  _Float16* H1   = (_Float16*)carve(S * 2);
  float*    KVb  = (float*)carve((size_t)BATCH * HEADS * DH * DH * 4);
  float*    KS   = (float*)carve((size_t)BATCH * DM * 4);
  float*    MEANB= (float*)carve((size_t)BATCH * DM * 4);
  float*    HD   = (float*)carve((size_t)MROWS * 64 * 4);
  float*    LOGA = (float*)carve((size_t)MROWS * 4);
  float*    LOGB = (float*)carve((size_t)MROWS * 4);
  float*    STP  = (float*)carve((size_t)BATCH * 4);

  /* f16 weight pool: 4 encoders x {Wq,Wk,Wv,Wo,W1,W2} + 2 head Wh1 */
  _Float16* wh[4][6];
  const EncW* encs[4] = { &rep, &fwd, &stop, &bwd };
  for (int e = 0; e < 4; ++e) {
    const float* src[6] = { encs[e]->Wq, encs[e]->Wk, encs[e]->Wv,
                            encs[e]->Wo, encs[e]->W1, encs[e]->W2 };
    for (int i = 0; i < 6; ++i) {
      wh[e][i] = (_Float16*)carve(65536 * 2);
      f2h_kernel<<<256, 256, 0, stream>>>(src[i], wh[e][i], 65536);
    }
  }
  _Float16* fwdWh1h = (_Float16*)carve(16384 * 2);
  _Float16* bwdWh1h = (_Float16*)carve(16384 * 2);
  f2h_kernel<<<64, 256, 0, stream>>>(fwdWh1, fwdWh1h, 16384);
  f2h_kernel<<<64, 256, 0, stream>>>(bwdWh1, bwdWh1h, 16384);

  const dim3 g256(MROWS / BM, DM / 128);  /* (1024, 2), BNT = 128 */
  const dim3 g64 (MROWS / BM, 1);         /* head1: N = 64, BNT = 64 */

  auto encoder = [&](const EncW& P, _Float16* const* W16,
                     const float* inF, const _Float16* inH,
                     float* outF, _Float16* outH) {
    gemm_kernel<128, ACT_ELU1, false, true,  false><<<g256, 256, 0, stream>>>(inH, W16[0], P.bq, nullptr, B0, nullptr, DM); /* Qf  */
    gemm_kernel<128, ACT_ELU1, false, true,  false><<<g256, 256, 0, stream>>>(inH, W16[1], P.bk, nullptr, B1, nullptr, DM); /* Kf  */
    gemm_kernel<128, ACT_NONE, false, true,  false><<<g256, 256, 0, stream>>>(inH, W16[2], P.bv, nullptr, B2, nullptr, DM); /* V   */
    ksum_kernel<<<BATCH, 256, 0, stream>>>(B1, KS);
    kv_kernel  <<<BATCH * HEADS, 256, 0, stream>>>(B1, B2, KVb);
    attn_kernel<<<BATCH * 16 * 4, 64, 0, stream>>>(B0, KVb, KS, H0);
    gemm_kernel<128, ACT_NONE, true,  true,  false><<<g256, 256, 0, stream>>>(H0, W16[3], P.bo, inF, B1, nullptr, DM);      /* +res */
    ln_kernel<<<MROWS, 256, 0, stream>>>(B1, P.g1, P.c1, B2, H0);
    gemm_kernel<128, ACT_RELU, false, false, true ><<<g256, 256, 0, stream>>>(H0, W16[4], P.f1, nullptr, nullptr, H1, DM);  /* FF1 */
    gemm_kernel<128, ACT_NONE, true,  true,  false><<<g256, 256, 0, stream>>>(H1, W16[5], P.f2, B2, B0, nullptr, DM);       /* FF2 */
    ln_kernel<<<MROWS, 256, 0, stream>>>(B0, P.g2, P.c2, B1, nullptr);
    ln_kernel<<<MROWS, 256, 0, stream>>>(B1, P.gf, P.cf, outF, outH);
  };

  /* -------- pipeline -------- */
  build_src_kernel<<<(unsigned)(S / 256), 256, 0, stream>>>(x, embed, SRCF, SRCH);

  encoder(rep, wh[0], SRCF, SRCH, REPF, REPH);

  /* backward branch */
  encoder(bwd, wh[3], REPF, REPH, B0, H0);
  gemm_kernel<64, ACT_LEAKY, false, true, false><<<g64, 256, 0, stream>>>(H0, bwdWh1h, bwdbh1, nullptr, HD, nullptr, 64);
  dot64_kernel<<<MROWS / 32, 256, 0, stream>>>(HD, bwdWh2, bwdbh2, LOGB);

  /* forward branch */
  encoder(fwd, wh[1], REPF, REPH, B0, H0);
  gemm_kernel<64, ACT_LEAKY, false, true, false><<<g64, 256, 0, stream>>>(H0, fwdWh1h, fwdbh1, nullptr, HD, nullptr, 64);
  dot64_kernel<<<MROWS / 32, 256, 0, stream>>>(HD, fwdWh2, fwdbh2, LOGA);

  /* stop branch */
  encoder(stop, wh[2], REPF, REPH, B0, H0);
  meanl_kernel<<<BATCH, 256, 0, stream>>>(B0, MEANB);
  stop_mlp_kernel<<<BATCH, 64, 0, stream>>>(MEANB, stopWh1, stopbh1, stopWh2, stopbh2, STP);

  /* outputs: forward_log_prob (128x1025) then backward_log_prob (128x1024) */
  float* out = (float*)d_out;
  lsm_kernel<<<BATCH, 256, 0, stream>>>(LOGA, STP, fmask, out, N2T + 1);
  lsm_kernel<<<BATCH, 256, 0, stream>>>(LOGB, nullptr, bmask, out + (size_t)BATCH * (N2T + 1), N2T);
}
